// AdditiveAttention_56865366999388
// MI455X (gfx1250) — matrix operand order repaired and checked by reading, not compiled, not run on hardware
//
#include <hip/hip_runtime.h>

// Problem constants (from reference): N=4, NQ=512, M=512, DQ=DK=DV=256, H=256
#define NB 4
#define NQS 512
#define MS 512
#define DDIM 256
#define HDIM 256
#define NSPLIT 4            // key-range splits per query tile (flash-style)
#define MSPAN 128           // MS / NSPLIT
#define NTILES (NB * (NQS / 16))   // 128 query tiles

typedef __attribute__((ext_vector_type(16))) _Float16 v16h;
typedef __attribute__((ext_vector_type(8)))  float    v8f;
typedef __attribute__((ext_vector_type(4)))  int      v4i;

__device__ __forceinline__ float fast_tanh(float x) {
#if __has_builtin(__builtin_amdgcn_tanhf)
    return __builtin_amdgcn_tanhf(x);          // v_tanh_f32 (gfx1250 trans pipe)
#elif __has_builtin(__builtin_amdgcn_tanh_f32)
    return __builtin_amdgcn_tanh_f32(x);
#else
    float e = __expf(x + x);                   // v_exp_f32 based fallback
    return 1.0f - 2.0f / (e + 1.0f);
#endif
}

// --- gfx1250 async memory->LDS path (ASYNCcnt), with safe fallback ----------
// Builtin signature (from hipcc diagnostic): (v4i AS1*, v4i AS3*, imm, imm)
#if __has_builtin(__builtin_amdgcn_global_load_async_to_lds_b128) && \
    __has_builtin(__builtin_amdgcn_s_wait_asynccnt)
#define USE_ASYNC_LDS 1
__device__ __forceinline__ void async_copy16(const float* g, float* l) {
    __builtin_amdgcn_global_load_async_to_lds_b128(
        (__attribute__((address_space(1))) v4i*)g,
        (__attribute__((address_space(3))) v4i*)l, 0, 0);
}
__device__ __forceinline__ void async_wait0() {
    __builtin_amdgcn_s_wait_asynccnt(0);
}
#else
#define USE_ASYNC_LDS 0
__device__ __forceinline__ void async_copy16(const float* g, float* l) {
    *(float4*)l = *(const float4*)g;           // VGPR round-trip fallback
}
__device__ __forceinline__ void async_wait0() {}
#endif

// -----------------------------------------------------------------------------
// Kernel 1: fused Q and K projection.  One wave computes one 16x16 f32 tile of
// qp = query @ W_q  (rows 0..2047)  or  kp = key @ W_k  (rows 2048..4095),
// K-loop over 256 in chunks of 32 via v_wmma_f32_16x16x32_f16.
// Fragment layouts follow CDNA5 ISA 7.12.2 (16-bit A 16x32, B 32x16, f32 C/D).
// -----------------------------------------------------------------------------
__global__ __launch_bounds__(32) void proj_kernel(
    const float* __restrict__ query, const float* __restrict__ key,
    const float* __restrict__ Wq,    const float* __restrict__ Wk,
    float* __restrict__ qp,          float* __restrict__ kp)
{
    const int lane    = threadIdx.x & 31;
    const int colTile = blockIdx.x;           // 0..15
    const int rbase   = blockIdx.y * 16;      // 0..4095 over q-rows then k-rows

    const float* X; const float* W; float* Y; int rloc;
    if (rbase < NB * NQS) { X = query; W = Wq; Y = qp; rloc = rbase; }
    else                  { X = key;   W = Wk; Y = kp; rloc = rbase - NB * NQS; }

    const int hl  = lane & 15;
    const int kg  = lane >> 4;
    const int row = rloc + hl;
    const int col = colTile * 16 + hl;
    const float* xr = X + row * DDIM;

    v8f c = {};
    for (int kk = 0; kk < DDIM; kk += 32) {
        v16h a, bfrag;
#pragma unroll
        for (int v = 0; v < 8; ++v) {
            const int k0 = kk + kg * 8 + ((v < 4) ? 2 * v : 16 + 2 * (v - 4));
            a[2 * v]     = (_Float16)xr[k0];
            a[2 * v + 1] = (_Float16)xr[k0 + 1];
        }
#pragma unroll
        for (int j = 0; j < 8; ++j) {
            const int k0 = kk + kg * 8 + ((j < 4) ? 2 * j : 16 + 2 * (j - 4));
            bfrag[2 * j]     = (_Float16)W[k0 * HDIM + col];
            bfrag[2 * j + 1] = (_Float16)W[(k0 + 1) * HDIM + col];
        }
        c = __builtin_amdgcn_wmma_f32_16x16x32_f16(false, a, false, bfrag,
                                                   (short)0, c, false, false);
    }
#pragma unroll
    for (int rr = 0; rr < 8; ++rr)
        Y[(rloc + rr + 8 * kg) * HDIM + col] = c[rr];
}

// -----------------------------------------------------------------------------
// Kernel 2: flash-style partial attention.  One block = (batch, 16-query tile,
// key-split of 128).  Computes tanh scores for its key range, local softmax
// stats (max, sum), and the unnormalized partial  P = exp(s - mx) @ V_range
// via WMMA.  Partial tile + stats go to workspace for the combine kernel.
// 256 threads (8 waves); each thread owns (1 query row x 2 key rows) per chunk.
// -----------------------------------------------------------------------------
__global__ __launch_bounds__(256) void attn_partial_kernel(
    const float* __restrict__ qp, const float* __restrict__ kp,
    const float* __restrict__ value, const float* __restrict__ Wv,
    float* __restrict__ P, float* __restrict__ mxB, float* __restrict__ sumB)
{
    __shared__ alignas(16) float    sQ[16][260];   // +4 pad: bank = 4n+h
    __shared__ alignas(16) float    sK[32][260];
    __shared__ alignas(16) _Float16 sA[16][136];   // exp(s-mx) as f16, K=128
    __shared__ float                sRed[16][17];
    __shared__ alignas(16) float    sWv[HDIM];

    const int tid = threadIdx.x;
    const int sp  = blockIdx.x & 3;                // key split 0..3
    const int nt  = (blockIdx.x >> 2) & 31;        // query tile 0..31
    const int b   = blockIdx.x >> 7;               // batch 0..3
    const int n0  = nt * 16;
    const int m0  = sp * MSPAN;

    sWv[tid] = Wv[tid];
    const float* qbase = qp + (b * NQS + n0) * HDIM;
    for (int i = tid; i < 16 * 64; i += 256) {     // 16 rows x 64 float4
        const int r = i >> 6, c4 = i & 63;
        async_copy16(&qbase[r * HDIM + c4 * 4], &sQ[r][c4 * 4]);
    }
    async_wait0();
    __syncthreads();

    const int n  = tid & 15;                       // query row within tile
    const int ml = tid >> 4;                       // key row 0..15 (owns ml, ml+16)
    const float* kb = kp + (b * MS + m0) * HDIM;
    float sc[8];                                   // 4 chunks x 2 key rows

#pragma unroll
    for (int mc = 0; mc < 4; ++mc) {
        for (int i = tid; i < 32 * 64; i += 256) { // 32 rows x 64 float4
            const int r = i >> 6, c4 = i & 63;
            async_copy16(&kb[(mc * 32 + r) * HDIM + c4 * 4], &sK[r][c4 * 4]);
        }
        if (mc < 3) __builtin_prefetch(&kb[((mc + 1) * 32 + ml) * HDIM], 0, 1);
        async_wait0();
        __syncthreads();
        float a0 = 0.f, a1 = 0.f;
#pragma unroll 4
        for (int h = 0; h < HDIM; h += 4) {        // 4 ds_b128 -> 8 tanh
            const float4 qv = *(const float4*)&sQ[n][h];
            const float4 wv = *(const float4*)&sWv[h];
            const float4 k0 = *(const float4*)&sK[ml][h];
            const float4 k1 = *(const float4*)&sK[ml + 16][h];
            a0 += wv.x * fast_tanh(qv.x + k0.x);
            a1 += wv.x * fast_tanh(qv.x + k1.x);
            a0 += wv.y * fast_tanh(qv.y + k0.y);
            a1 += wv.y * fast_tanh(qv.y + k1.y);
            a0 += wv.z * fast_tanh(qv.z + k0.z);
            a1 += wv.z * fast_tanh(qv.z + k1.z);
            a0 += wv.w * fast_tanh(qv.w + k0.w);
            a1 += wv.w * fast_tanh(qv.w + k1.w);
        }
        sc[2 * mc]     = a0;
        sc[2 * mc + 1] = a1;
        __syncthreads();                           // sK reuse next chunk
    }

    // ---- local softmax stats over this block's 128 keys ----
    float mx = -3.0e38f;
#pragma unroll
    for (int i = 0; i < 8; ++i) mx = fmaxf(mx, sc[i]);
    sRed[n][ml] = mx;
    __syncthreads();
#pragma unroll
    for (int t = 0; t < 16; ++t) mx = fmaxf(mx, sRed[n][t]);
    __syncthreads();
    float sum = 0.f;
#pragma unroll
    for (int i = 0; i < 8; ++i) {
        const float e = __expf(sc[i] - mx);
        sc[i] = e;
        sum += e;
    }
    sRed[n][ml] = sum;
    __syncthreads();
    sum = 0.f;
#pragma unroll
    for (int t = 0; t < 16; ++t) sum += sRed[n][t];

    const int tileIdx = (b * 32 + nt) * NSPLIT + sp;
    if (tid < 16) {                                // n == tid here
        mxB [tileIdx * 16 + tid] = mx;
        sumB[tileIdx * 16 + tid] = sum;
    }
#pragma unroll
    for (int mc = 0; mc < 4; ++mc) {               // unnormalized E as f16
        sA[n][mc * 32 + ml]      = (_Float16)sc[2 * mc];
        sA[n][mc * 32 + ml + 16] = (_Float16)sc[2 * mc + 1];
    }
    __syncthreads();

    // ---- partial GEMM: P(16x256) = E(16x128) @ V[m0:m0+128](128x256) ----
    const int lane = tid & 31;
    const int wav  = tid >> 5;                     // 8 waves x 2 col-tiles
    const int hl   = lane & 15;
    const int kg   = lane >> 4;
    const float* vb = value + (b * MS + m0) * DDIM;
    float* pb = P + tileIdx * (16 * DDIM);

#pragma unroll
    for (int cti = 0; cti < 2; ++cti) {
        const int col = (wav + cti * 8) * 16 + hl;
        v8f c = {};
        for (int kk = 0; kk < MSPAN; kk += 32) {
            v16h a, bfrag;
#pragma unroll
            for (int v = 0; v < 8; ++v) {
                const int k0 = kk + kg * 8 + ((v < 4) ? 2 * v : 16 + 2 * (v - 4));
                a[2 * v]     = sA[hl][k0];
                a[2 * v + 1] = sA[hl][k0 + 1];
            }
#pragma unroll
            for (int j = 0; j < 8; ++j) {
                const int k0 = kk + kg * 8 + ((j < 4) ? 2 * j : 16 + 2 * (j - 4));
                bfrag[2 * j]     = (_Float16)vb[k0 * DDIM + col];
                bfrag[2 * j + 1] = (_Float16)vb[(k0 + 1) * DDIM + col];
            }
            c = __builtin_amdgcn_wmma_f32_16x16x32_f16(false, a, false, bfrag,
                                                       (short)0, c, false, false);
        }
#pragma unroll
        for (int rr = 0; rr < 8; ++rr)
            pb[(rr + 8 * kg) * DDIM + col] = c[rr];
    }
}

// -----------------------------------------------------------------------------
// Kernel 3: combine the NSPLIT partials with log-sum-exp rescaling:
//   out[r][c] = sum_j e^{mx_j - M} * P_j[r][c]  /  sum_j e^{mx_j - M} * sum_j
// -----------------------------------------------------------------------------
__global__ __launch_bounds__(256) void combine_kernel(
    const float* __restrict__ P, const float* __restrict__ mxB,
    const float* __restrict__ sumB, float* __restrict__ out)
{
    __shared__ float coef[NSPLIT][16];
    const int tid = threadIdx.x;
    const int nt  = blockIdx.x & 31;
    const int b   = blockIdx.x >> 5;
    const int tbase = (b * 32 + nt) * NSPLIT;

    if (tid < 16) {
        const int r = tid;
        float mxv[NSPLIT], M = -3.0e38f;
#pragma unroll
        for (int j = 0; j < NSPLIT; ++j) {
            mxv[j] = mxB[(tbase + j) * 16 + r];
            M = fmaxf(M, mxv[j]);
        }
        float den = 0.f, cj[NSPLIT];
#pragma unroll
        for (int j = 0; j < NSPLIT; ++j) {
            cj[j] = __expf(mxv[j] - M);
            den += cj[j] * sumB[(tbase + j) * 16 + r];
        }
        const float inv = 1.0f / den;
#pragma unroll
        for (int j = 0; j < NSPLIT; ++j) coef[j][r] = cj[j] * inv;
    }
    __syncthreads();

    const float* pb = P + tbase * (16 * DDIM);
    float* ob = out + (b * NQS + nt * 16) * DDIM;
    for (int i = tid; i < 16 * DDIM; i += 256) {
        const int r = i >> 8;                      // DDIM == 256
        float acc = 0.f;
#pragma unroll
        for (int j = 0; j < NSPLIT; ++j)
            acc += coef[j][r] * pb[j * (16 * DDIM) + i];
        ob[i] = acc;
    }
}

// -----------------------------------------------------------------------------
extern "C" void kernel_launch(void* const* d_in, const int* in_sizes, int n_in,
                              void* d_out, int out_size, void* d_ws, size_t ws_size,
                              hipStream_t stream) {
    const float* query = (const float*)d_in[0];   // (4, 512, 256)
    const float* key   = (const float*)d_in[1];   // (4, 512, 256)
    const float* value = (const float*)d_in[2];   // (4, 512, 256)
    const float* Wq    = (const float*)d_in[3];   // (256, 256)
    const float* Wk    = (const float*)d_in[4];   // (256, 256)
    const float* Wv    = (const float*)d_in[5];   // (256,)
    float* out = (float*)d_out;                   // (4, 512, 256)

    float* qp   = (float*)d_ws;                        // 2 MB
    float* kp   = qp + NB * NQS * HDIM;                // 2 MB
    float* P    = kp + NB * MS * HDIM;                 // 512 tiles * 16*256 f32 = 8 MB
    float* mxB  = P + NTILES * NSPLIT * 16 * DDIM;     // 32 KB
    float* sumB = mxB + NTILES * NSPLIT * 16;          // 32 KB

    // projections: 4096 one-wave tiles (q-rows then k-rows)
    dim3 g1(16, 256);
    proj_kernel<<<g1, 32, 0, stream>>>(query, key, Wq, Wk, qp, kp);

    // 4 batches x 32 query tiles x 4 key splits = 512 blocks
    attn_partial_kernel<<<NTILES * NSPLIT, 256, 0, stream>>>(
        qp, kp, value, Wv, P, mxB, sumB);

    // merge partials: 128 blocks
    combine_kernel<<<NTILES, 256, 0, stream>>>(P, mxB, sumB, out);
}
